// sLSTMCell_45921790328950
// MI455X (gfx1250) — compile-verified, matrix operand order checked by reading, not run on hardware
//
#include <hip/hip_runtime.h>
#include <math.h>

// ---------------------------------------------------------------------------
// sLSTM cell, B=8192, D=1024, H=1024.
//   pre = [x,h] @ [W;R] + b   (single GEMM: [8192,2048] x [2048,4096])
//   gates -> h_t, c_t, n_t  (fused into GEMM epilogue)
// bf16 WMMA GEMM with Tensor-Data-Mover (TDM) double-buffered LDS staging.
// ---------------------------------------------------------------------------

typedef __attribute__((ext_vector_type(16))) __bf16 v16bf;
typedef __attribute__((ext_vector_type(8)))  __bf16 v8bf;
typedef __attribute__((ext_vector_type(8)))  float  v8f;
typedef __attribute__((ext_vector_type(4)))  unsigned int u32x4;
typedef __attribute__((ext_vector_type(4)))  int          i32x4;
typedef __attribute__((ext_vector_type(8)))  int          i32x8;

#define BDIM 8192
#define DDIM 1024
#define HDIM 1024
#define KDIM 2048          // D + H
#define NDIM 4096          // 4*H

#define BM 128
#define BN 32              // per-gate N tile; block covers 4 gates
#define KT 32
#define KTILES (KDIM / KT) // 64

#if defined(__gfx1250__) && __has_builtin(__builtin_amdgcn_tensor_load_to_lds) && \
    __has_builtin(__builtin_amdgcn_s_wait_tensorcnt)
#define USE_TDM 1
#else
#define USE_TDM 0
#endif

static __device__ __forceinline__ __bf16 f2bf(float f) {
    union { float f; unsigned u; } a; a.f = f;
    unsigned r = a.u + 0x7fffu + ((a.u >> 16) & 1u);   // round-nearest-even
    unsigned short h = (unsigned short)(r >> 16);
    union { unsigned short s; __bf16 b; } o; o.s = h;
    return o.b;
}

// ---------------------------------------------------------------------------
// Pack A = [x | h] -> bf16 [8192][2048], 8 elements per thread (16B stores)
// ---------------------------------------------------------------------------
__global__ __launch_bounds__(256) void pack_a(const float* __restrict__ x,
                                              const float* __restrict__ h,
                                              __bf16* __restrict__ A) {
    long i = ((long)blockIdx.x * blockDim.x + threadIdx.x) * 8;  // element idx
    long row  = i >> 11;           // / 2048
    long colb = i & 2047;
    const float* src = (colb < DDIM) ? (x + row * (long)DDIM + colb)
                                     : (h + row * (long)HDIM + (colb - DDIM));
    v8bf v;
#pragma unroll
    for (int j = 0; j < 8; ++j) v[j] = f2bf(src[j]);
    *(v8bf*)(A + i) = v;
}

// ---------------------------------------------------------------------------
// Pack Bt[n][k] = (k<1024 ? W[k][n] : R[k-1024][n]) -> bf16 [4096][2048]
// LDS-tiled 32x32 transpose, coalesced reads and writes.
// ---------------------------------------------------------------------------
__global__ __launch_bounds__(256) void pack_b(const float* __restrict__ W,
                                              const float* __restrict__ R,
                                              __bf16* __restrict__ Bt) {
    __shared__ __bf16 tile[32][33];
    const int k0 = blockIdx.x * 32;
    const int n0 = blockIdx.y * 32;
    const int tx = threadIdx.x & 31;
    const int ty = threadIdx.x >> 5;        // 0..7
#pragma unroll
    for (int r = 0; r < 32; r += 8) {
        int k = k0 + ty + r;
        float v = (k < DDIM) ? W[(long)k * NDIM + n0 + tx]
                             : R[(long)(k - DDIM) * NDIM + n0 + tx];
        tile[tx][ty + r] = f2bf(v);
    }
    __syncthreads();
#pragma unroll
    for (int r = 0; r < 32; r += 8) {
        int n = n0 + ty + r;
        Bt[(long)n * KDIM + k0 + tx] = tile[ty + r][tx];
    }
}

// ---------------------------------------------------------------------------
// TDM descriptor helper: load a (tile_d2 x tile_d1 x tile_d0) bf16 tile to
// LDS with hardware row-padding (pad_interval=3 -> every 64B; pad_amount=3 ->
// 16B), giving an 80B LDS row pitch (matches the 5-uint4 fragment layout).
// D# packing per CDNA5 ISA ch.8 (groups 0..3).
// ---------------------------------------------------------------------------
#if USE_TDM
static __device__ __forceinline__ void tdm_load_tile(
        unsigned lds_off, const void* gptr,
        unsigned tile_d0, unsigned tile_d1, unsigned tile_d2,
        unsigned long long s0,        // tensor_dim0_stride (elements)
        unsigned long long s1,        // tensor_dim1_stride (elements)
        unsigned t0, unsigned t1, unsigned t2)
{
    const unsigned long long ga = (unsigned long long)gptr;
    u32x4 g0;
    g0[0] = 1u;                                        // count=1 (valid), user
    g0[1] = lds_off;                                   // lds_addr (bytes)
    g0[2] = (unsigned)(ga & 0xffffffffu);              // global_addr[31:0]
    g0[3] = (unsigned)((ga >> 32) & 0x1ffffffu)        // global_addr[56:32]
          | (2u << 30);                                // type=2 (image)
    i32x8 g1;
    // data_size=1 (2B), pad_enable=1, pad_interval=3 (16 DW), pad_amount=3 (4 DW)
    g1[0] = (int)((1u << 16) | (1u << 20) | (3u << 22) | (3u << 25));
    g1[1] = (int)((t0 & 0xffffu) << 16);                               // dim0 lo
    g1[2] = (int)(((t0 >> 16) & 0xffffu) | ((t1 & 0xffffu) << 16));    // dim0 hi | dim1 lo
    g1[3] = (int)(((t1 >> 16) & 0xffffu) | ((tile_d0 & 0xffffu) << 16));
    g1[4] = (int)((tile_d1 & 0xffffu) | ((tile_d2 & 0xffffu) << 16));
    g1[5] = (int)(s0 & 0xffffffffu);
    g1[6] = (int)(((s0 >> 32) & 0xffffu) | ((s1 & 0xffffu) << 16));
    g1[7] = (int)((s1 >> 16) & 0xffffffffu);
    i32x4 g2;
    g2[0] = (int)t2;            // tensor_dim2
    g2[1] = 0;                  // tensor_dim3
    g2[2] = 0;                  // tensor_dim2_stride[31:0]
    g2[3] = 0;                  // stride hi | tile_dim3=0
    i32x4 g3 = {0, 0, 0, 0};
#if __clang_major__ >= 23
    i32x8 g4 = {0, 0, 0, 0, 0, 0, 0, 0};
    __builtin_amdgcn_tensor_load_to_lds(g0, g1, g2, g3, g4, 0);
#else
    __builtin_amdgcn_tensor_load_to_lds(g0, g1, g2, g3, 0);
#endif
}
#endif

// ---------------------------------------------------------------------------
// Fused GEMM + sLSTM gates.
// Block: 256 threads (8 waves). Tile: BM=128 rows x (4 gates x BN=32 cols).
// Wave w owns rows [m0 + 16w, m0 + 16w + 16); 8 accumulators (4 gates x 2
// 16-wide n-subtiles). LDS rows pitched 80B (5 x uint4) to spread banks.
// Staging: wave 0 issues TDM tensor_load_to_lds for the next k-tile while all
// waves compute the current one; sync via s_wait_tensorcnt + barrier.
// ---------------------------------------------------------------------------
union FragU { uint4 u[2]; v16bf v; };

__global__ __launch_bounds__(256) void slstm_fused(
        const __bf16* __restrict__ A,      // [8192][2048]
        const __bf16* __restrict__ Bt,     // [4096][2048]
        const float*  __restrict__ bias,   // [4096] (z,i,f,o)
        const float*  __restrict__ c_prev, // [8192][1024]
        const float*  __restrict__ n_prev, // [8192][1024]
        float* __restrict__ out)           // h_t | c_t | n_t, each 8192*1024
{
    __shared__ uint4 sA[2][BM * 5];       // 128 rows x 4 chunks (+16B pad)
    __shared__ uint4 sB[2][BM * 5];       // 4*32 rows x 4 chunks (+16B pad)

    const int tid  = threadIdx.x;
    const int wave = tid >> 5;
    const int lane = tid & 31;
    const int half = lane >> 4;           // K-half selector per ISA 7.12.2
    const int col  = lane & 15;

    const int m0 = blockIdx.x * BM;
    const int n0 = blockIdx.y * BN;

    v8f acc[4][2];
#pragma unroll
    for (int g = 0; g < 4; ++g)
#pragma unroll
        for (int t = 0; t < 2; ++t) acc[g][t] = (v8f)0.0f;

#if USE_TDM
    // -------- TDM staging: one DMA per (buffer, operand) issued by wave 0 ----
    const unsigned ldsA[2] = { (unsigned)(unsigned long long)(&sA[0][0]),
                               (unsigned)(unsigned long long)(&sA[1][0]) };
    const unsigned ldsB[2] = { (unsigned)(unsigned long long)(&sB[0][0]),
                               (unsigned)(unsigned long long)(&sB[1][0]) };
    const __bf16* gA = A  + (long)m0 * KDIM;     // + k0 per tile
    const __bf16* gB = Bt + (long)n0 * KDIM;     // rows n0+g*1024, + k0 per tile

    if (wave == 0) {
        // A tile: 2D, 128 rows x 32 (K-contiguous)
        tdm_load_tile(ldsA[0], gA, KT, BM, 0,
                      KDIM, 0, KDIM, BDIM, 1);
        // B tile: 3D, 4 gate panels x 32 rows x 32, Z-stride = H*KDIM
        tdm_load_tile(ldsB[0], gB, KT, BN, 4,
                      KDIM, (unsigned long long)HDIM * KDIM, KDIM, NDIM, 4);
        __builtin_amdgcn_s_wait_tensorcnt(0);
    }
    __syncthreads();
#else
    // -------- fallback staging: per-thread b128 loads + ds_store ------------
    int c0r[2], c0k[2], brow[2];
#pragma unroll
    for (int i = 0; i < 2; ++i) {
        int c   = tid + i * 256;
        c0r[i]  = c >> 2;
        c0k[i]  = c & 3;
        brow[i] = (c0r[i] >> 5) * HDIM + n0 + (c0r[i] & 31);
    }
#pragma unroll
    for (int i = 0; i < 2; ++i) {
        sA[0][c0r[i] * 5 + c0k[i]] =
            *(const uint4*)(A + (long)(m0 + c0r[i]) * KDIM + c0k[i] * 8);
        sB[0][c0r[i] * 5 + c0k[i]] =
            *(const uint4*)(Bt + (long)brow[i] * KDIM + c0k[i] * 8);
    }
    __syncthreads();
#endif

    const int arow = wave * 16 + col;

    for (int kt = 0; kt < KTILES; ++kt) {
        const int cur = kt & 1;
        const int nxt = cur ^ 1;

        if (kt + 1 < KTILES) {
            const long k0 = (long)(kt + 1) * KT;
#if USE_TDM
            if (wave == 0) {
                tdm_load_tile(ldsA[nxt], gA + k0, KT, BM, 0,
                              KDIM, 0, KDIM, BDIM, 1);
                tdm_load_tile(ldsB[nxt], gB + k0, KT, BN, 4,
                              KDIM, (unsigned long long)HDIM * KDIM, KDIM, NDIM, 4);
            }
#else
#pragma unroll
            for (int i = 0; i < 2; ++i) {
                sA[nxt][c0r[i] * 5 + c0k[i]] =
                    *(const uint4*)(A + (long)(m0 + c0r[i]) * KDIM + k0 + c0k[i] * 8);
                sB[nxt][c0r[i] * 5 + c0k[i]] =
                    *(const uint4*)(Bt + (long)brow[i] * KDIM + k0 + c0k[i] * 8);
            }
            if (kt + 2 < KTILES) {
                const long kp = (long)(kt + 2) * KT;
                __builtin_prefetch(A + (long)(m0 + c0r[0]) * KDIM + kp, 0, 1);
                __builtin_prefetch(Bt + (long)brow[0] * KDIM + kp, 0, 1);
            }
#endif
        }

        // A fragment: 16x32 bf16, lane layout per ISA 7.12.2
        FragU fa;
        fa.u[0] = sA[cur][arow * 5 + half];        // K  0..15 half
        fa.u[1] = sA[cur][arow * 5 + 2 + half];    // K 16..31 half
#pragma unroll
        for (int g = 0; g < 4; ++g) {
#pragma unroll
            for (int t = 0; t < 2; ++t) {
                FragU fb;
                const int br = g * BN + t * 16 + col;
                fb.u[0] = sB[cur][br * 5 + half];
                fb.u[1] = sB[cur][br * 5 + 2 + half];
                acc[g][t] = __builtin_amdgcn_wmma_f32_16x16x32_bf16(
                    false, fa.v, false, fb.v, (short)0, acc[g][t], false, false);
            }
        }
#if USE_TDM
        if (wave == 0) __builtin_amdgcn_s_wait_tensorcnt(0);
#endif
        __syncthreads();
    }

    // ------------------- fused sLSTM gate epilogue -------------------------
    float* __restrict__ hO = out;
    float* __restrict__ cO = out + (long)BDIM * HDIM;
    float* __restrict__ nO = out + 2L * BDIM * HDIM;
    const int m_base = m0 + wave * 16;

#pragma unroll
    for (int t = 0; t < 2; ++t) {
        const int n = n0 + t * 16 + col;
        const float bZ = bias[n];
        const float bI = bias[HDIM + n];
        const float bF = bias[2 * HDIM + n];
        const float bO = bias[3 * HDIM + n];
#pragma unroll
        for (int e = 0; e < 8; ++e) {
            const int  m   = m_base + e + half * 8;   // C/D layout: VGPR e
            const long idx = (long)m * HDIM + n;
            const float pz = acc[0][t][e] + bZ;
            const float pi = acc[1][t][e] + bI;
            const float pf = acc[2][t][e] + bF;
            const float po = acc[3][t][e] + bO;
            const float z  = tanhf(pz);
            const float iv = fminf(__expf(pi), 50.0f);
            const float f  = 1.0f / (1.0f + __expf(-pf));
            const float o  = 1.0f / (1.0f + __expf(-po));
            const float cv = f * c_prev[idx] + iv * z;
            const float nv = f * n_prev[idx] + iv;
            const float hv = o * (cv / (nv + 1e-6f));
            hO[idx] = hv;
            cO[idx] = cv;
            nO[idx] = nv;
        }
    }
}

// ---------------------------------------------------------------------------
extern "C" void kernel_launch(void* const* d_in, const int* in_sizes, int n_in,
                              void* d_out, int out_size, void* d_ws, size_t ws_size,
                              hipStream_t stream) {
    const float* x_t    = (const float*)d_in[0];
    const float* h_prev = (const float*)d_in[1];
    const float* c_prev = (const float*)d_in[2];
    const float* n_prev = (const float*)d_in[3];
    const float* W      = (const float*)d_in[4];
    const float* R      = (const float*)d_in[5];
    const float* b      = (const float*)d_in[6];
    float* out = (float*)d_out;

    // workspace: A bf16 (32MB) + Bt bf16 (16MB)
    __bf16* Abf = (__bf16*)d_ws;
    __bf16* Btb = Abf + (size_t)BDIM * KDIM;

    // 1) pack activations: 8192*2048 / 8 per thread = 2M threads
    pack_a<<<dim3((BDIM * KDIM / 8) / 256), dim3(256), 0, stream>>>(x_t, h_prev, Abf);

    // 2) pack + transpose weights: tiles of 32x32 over [2048 x 4096]
    pack_b<<<dim3(KDIM / 32, NDIM / 32), dim3(256), 0, stream>>>(W, R, Btb);

    // 3) fused GEMM + gates: grid 64 x 32
    slstm_fused<<<dim3(BDIM / BM, HDIM / BN), dim3(256), 0, stream>>>(
        Abf, Btb, b, c_prev, n_prev, out);
}
